// GVPEncoder_58256936403256
// MI455X (gfx1250) — matrix operand order repaired and checked by reading, not compiled, not run on hardware
//
#include <hip/hip_runtime.h>
#include <math.h>

// ---------------- problem constants ----------------
#define NN 30000
#define NE 480000
#define SI 128
#define VI 16
#define SE 32
#define SMI 288      // 2*SI + SE
#define VMI 33       // 2*VI + 1
#define H1 33
#define K1T 11       // ceil(321/32) -> K padded to 352
#define K2T 5        // ceil(144/32) -> 160
#define F1KT 5       // FF1: K = 160
#define F2KT 17      // FF2: K = 544
#define XSTR 368     // edge X row stride (bf16), 736B = 46*16B
#define X2STR 704    // node X row stride (bf16), 1408B = 88*16B

// packed-weight offsets in bf16 elements (each tile = 512 bf16)
#define W1_OFF 0                 // 8*11 = 88 tiles
#define W2_OFF (88 * 512)        // 8*5  = 40 tiles
#define W3_OFF (128 * 512)       // 8*5  = 40 tiles
#define F1_OFF (168 * 512)       // 32*5 = 160 tiles
#define F2_OFF (328 * 512)       // 8*17 = 136 tiles
#define WPK_ELEMS (464 * 512)
#define WPK_BYTES (WPK_ELEMS * 2)

typedef __attribute__((ext_vector_type(16))) __bf16 v16bf;
typedef __attribute__((ext_vector_type(8)))  __bf16 v8bf;
typedef __attribute__((ext_vector_type(8)))  float  v8f;

__device__ __forceinline__ __bf16 f2bf(float f) {
  union { float f; unsigned u; } a; a.f = f;
  unsigned u = a.u;
  u += 0x7FFFu + ((u >> 16) & 1u);   // round to nearest even
  unsigned short h = (unsigned short)(u >> 16);
  __bf16 r;
  __builtin_memcpy(&r, &h, 2);
  return r;
}

// C[16x16] += X[16 x 32*ktiles] * W^T ; X staged in LDS, W pre-packed fragments.
// A layout (16-bit 16x32): lane m (0-15): elems 0-7 -> K 0-7, 8-15 -> K 16-23;
// lanes 16-31: K 8-15 / 24-31.
__device__ __forceinline__ v8f wmma_acc_tile(const __bf16* Xbase, int xstride,
                                             const __bf16* Wt, int ktiles, int lane) {
  const int row = lane & 15, hi = lane >> 4;
  v8f acc = {};
  for (int kt = 0; kt < ktiles; ++kt) {
    const __bf16* ap = Xbase + row * xstride + kt * 32 + 8 * hi;
    v8bf lo = *(const v8bf*)ap;
    v8bf hp = *(const v8bf*)(ap + 16);
    v16bf a;
#pragma unroll
    for (int i = 0; i < 8; ++i) { a[i] = lo[i]; a[i + 8] = hp[i]; }
    v16bf b = *(const v16bf*)(Wt + ((size_t)kt * 32 + lane) * 16);
    acc = __builtin_amdgcn_wmma_f32_16x16x32_bf16(false, a, false, b,
                                                  (short)0, acc, false, false);
  }
  return acc;
}

// Pack W[No x Ki] (row major, f32) into per-tile B fragments.
// B layout (16-bit 32x16): lane l: column n = l&15; lanes 0-15 hold K 0-7 / 16-23,
// lanes 16-31 hold K 8-15 / 24-31 of the 32-row tile (matches A layout).
__global__ void pack_w(const float* __restrict__ W, __bf16* __restrict__ dst,
                       int No, int Ki, int ktiles, int ntiles) {
  int idx = blockIdx.x * 256 + threadIdx.x;
  int total = ntiles * ktiles * 512;
  if (idx >= total) return;
  int tile = idx >> 9, r = idx & 511;
  int lane = r >> 4, e = r & 15;
  int nt = tile / ktiles, kt = tile - nt * ktiles;
  int n = nt * 16 + (lane & 15);
  int k = kt * 32 + (lane >> 4) * 8 + (e < 8 ? e : e + 8);
  float val = (n < No && k < Ki) ? W[(size_t)n * Ki + k] : 0.f;
  dst[idx] = f2bf(val);
}

// =========================== edge (message) kernel ===========================
#define EKW 4
__global__ __launch_bounds__(128) void edge_kernel(
    const float* __restrict__ s, const float* __restrict__ v,
    const int* __restrict__ eidx,
    const float* __restrict__ edge_s, const float* __restrict__ edge_v,
    const float* __restrict__ m1_wh, const float* __restrict__ m1_wsb, const float* __restrict__ m1_wv,
    const float* __restrict__ m2_wh, const float* __restrict__ m2_wsb, const float* __restrict__ m2_wv,
    const float* __restrict__ m3_wh, const float* __restrict__ m3_wsb, const float* __restrict__ m3_wv,
    const __bf16* __restrict__ wpk,
    float* __restrict__ agg_s, float* __restrict__ agg_v, float* __restrict__ cnt) {
  __shared__ __align__(16) __bf16 Xs[EKW][16][XSTR];
  __shared__ float MV[EKW][16][100];
  __shared__ float VH[EKW][16][100];
  __shared__ float VA[EKW][16][48];
  __shared__ int SID[EKW][16], DID[EKW][16];
  __shared__ float w_h1[H1 * VMI];
  __shared__ float w_v1[VI * H1];
  __shared__ float w_h2[VI * VI], w_v2[VI * VI], w_h3[VI * VI], w_v3[VI * VI];
  __shared__ float b1[SI], b2[SI], b3[SI];

  const int tid = threadIdx.x;
  for (int i = tid; i < H1 * VMI; i += 128) w_h1[i] = m1_wh[i];
  for (int i = tid; i < VI * H1; i += 128) w_v1[i] = m1_wv[i];
  for (int i = tid; i < VI * VI; i += 128) {
    w_h2[i] = m2_wh[i]; w_v2[i] = m2_wv[i];
    w_h3[i] = m3_wh[i]; w_v3[i] = m3_wv[i];
  }
  for (int i = tid; i < SI; i += 128) { b1[i] = m1_wsb[i]; b2[i] = m2_wsb[i]; b3[i] = m3_wsb[i]; }

  const int w = tid >> 5, lane = tid & 31;
  const int grp = blockIdx.x * EKW + w;
  const bool ok = grp < (NE / 16);
  const int e0 = grp * 16;
  const int row = lane & 15, hi = lane >> 4;
  if (ok && lane < 16) {
    SID[w][lane] = eidx[e0 + lane];
    DID[w][lane] = eidx[NE + e0 + lane];
  }
  __syncthreads();

  // ---- gather concatenated features ----
  if (ok) {
    for (int i = lane; i < 16 * SMI; i += 32) {
      int r = i / SMI, c = i - r * SMI;
      float val;
      if (c < SI)           val = s[(size_t)SID[w][r] * SI + c];
      else if (c < SI + SE) val = edge_s[(size_t)(e0 + r) * SE + (c - SI)];
      else                  val = s[(size_t)DID[w][r] * SI + (c - SI - SE)];
      Xs[w][r][c] = f2bf(val);
    }
    for (int i = lane; i < 16 * 47; i += 32) {  // zero K pad 321..367
      int r = i / 47;
      Xs[w][r][321 + (i - r * 47)] = f2bf(0.f);
    }
    for (int i = lane; i < 16 * VMI * 3; i += 32) {
      int r = i / (VMI * 3), rem = i - r * (VMI * 3), c = rem / 3, t = rem - c * 3;
      float val;
      if (c < VI)      val = v[((size_t)SID[w][r] * VI + c) * 3 + t];
      else if (c == VI) val = edge_v[(size_t)(e0 + r) * 3 + t];
      else             val = v[((size_t)DID[w][r] * VI + (c - VI - 1)) * 3 + t];
      MV[w][r][c * 3 + t] = val;
    }
  }
  __syncthreads();

  // ---- GVP1: vh1 = wh1 @ mv ; vn1 -> X cols 288..320 ----
  if (ok) {
    for (int i = lane; i < 16 * H1 * 3; i += 32) {
      int r = i / (H1 * 3), rem = i - r * (H1 * 3), h = rem / 3, t = rem - h * 3;
      float a = 0.f;
      for (int c = 0; c < VMI; ++c) a += w_h1[h * VMI + c] * MV[w][r][c * 3 + t];
      VH[w][r][h * 3 + t] = a;
    }
  }
  __syncthreads();
  if (ok) {
    for (int i = lane; i < 16 * H1; i += 32) {
      int r = i / H1, h = i - r * H1;
      float x = VH[w][r][h * 3], y = VH[w][r][h * 3 + 1], z = VH[w][r][h * 3 + 2];
      Xs[w][r][SMI + h] = f2bf(sqrtf(fmaxf(x * x + y * y + z * z, 1e-8f)));
    }
  }
  __syncthreads();

  v8f acc[8];
  // ---- GVP1 scalar GEMM + gated vector out ----
  if (ok) {
    for (int nt = 0; nt < 8; ++nt)
      acc[nt] = wmma_acc_tile(&Xs[w][0][0], XSTR, wpk + W1_OFF + (size_t)nt * K1T * 512, K1T, lane);
    for (int i = lane; i < 16 * VI; i += 32) {
      int r = i >> 4, o = i & 15;
      float ox = 0, oy = 0, oz = 0;
      for (int h = 0; h < H1; ++h) {
        float wv = w_v1[o * H1 + h];
        ox += wv * VH[w][r][h * 3]; oy += wv * VH[w][r][h * 3 + 1]; oz += wv * VH[w][r][h * 3 + 2];
      }
      float n = sqrtf(fmaxf(ox * ox + oy * oy + oz * oz, 1e-8f));
      float g = 1.f / (1.f + expf(-n));
      VA[w][r][o * 3] = ox * g; VA[w][r][o * 3 + 1] = oy * g; VA[w][r][o * 3 + 2] = oz * g;
    }
  }
  __syncthreads();
  // s1=relu(.) -> X cols 0..127 ; zero 144..159 ; vh2 = wh2 @ v1
  if (ok) {
    for (int nt = 0; nt < 8; ++nt) {
#pragma unroll
      for (int q = 0; q < 8; ++q) {
        int m = q + 8 * hi, nc = nt * 16 + row;
        Xs[w][m][nc] = f2bf(fmaxf(acc[nt][q] + b1[nc], 0.f));
      }
    }
    for (int i = lane; i < 256; i += 32) Xs[w][i >> 4][144 + (i & 15)] = f2bf(0.f);
    for (int i = lane; i < 16 * 48; i += 32) {
      int r = i / 48, rem = i - r * 48, h = rem / 3, t = rem - h * 3;
      float a = 0.f;
      for (int c = 0; c < VI; ++c) a += w_h2[h * VI + c] * VA[w][r][c * 3 + t];
      VH[w][r][h * 3 + t] = a;
    }
  }
  __syncthreads();
  if (ok) {
    for (int i = lane; i < 256; i += 32) {
      int r = i >> 4, h = i & 15;
      float x = VH[w][r][h * 3], y = VH[w][r][h * 3 + 1], z = VH[w][r][h * 3 + 2];
      Xs[w][r][SI + h] = f2bf(sqrtf(fmaxf(x * x + y * y + z * z, 1e-8f)));
    }
  }
  __syncthreads();
  // ---- GVP2 ----
  if (ok) {
    for (int nt = 0; nt < 8; ++nt)
      acc[nt] = wmma_acc_tile(&Xs[w][0][0], XSTR, wpk + W2_OFF + (size_t)nt * K2T * 512, K2T, lane);
    for (int i = lane; i < 256; i += 32) {  // gated v2 -> VA
      int r = i >> 4, o = i & 15;
      float ox = 0, oy = 0, oz = 0;
      for (int h = 0; h < VI; ++h) {
        float wv = w_v2[o * VI + h];
        ox += wv * VH[w][r][h * 3]; oy += wv * VH[w][r][h * 3 + 1]; oz += wv * VH[w][r][h * 3 + 2];
      }
      float n = sqrtf(fmaxf(ox * ox + oy * oy + oz * oz, 1e-8f));
      float g = 1.f / (1.f + expf(-n));
      VA[w][r][o * 3] = ox * g; VA[w][r][o * 3 + 1] = oy * g; VA[w][r][o * 3 + 2] = oz * g;
    }
  }
  __syncthreads();
  if (ok) {
    for (int nt = 0; nt < 8; ++nt) {
#pragma unroll
      for (int q = 0; q < 8; ++q) {
        int m = q + 8 * hi, nc = nt * 16 + row;
        Xs[w][m][nc] = f2bf(fmaxf(acc[nt][q] + b2[nc], 0.f));
      }
    }
    for (int i = lane; i < 16 * 48; i += 32) {  // vh3
      int r = i / 48, rem = i - r * 48, h = rem / 3, t = rem - h * 3;
      float a = 0.f;
      for (int c = 0; c < VI; ++c) a += w_h3[h * VI + c] * VA[w][r][c * 3 + t];
      VH[w][r][h * 3 + t] = a;
    }
  }
  __syncthreads();
  if (ok) {
    for (int i = lane; i < 256; i += 32) {
      int r = i >> 4, h = i & 15;
      float x = VH[w][r][h * 3], y = VH[w][r][h * 3 + 1], z = VH[w][r][h * 3 + 2];
      Xs[w][r][SI + h] = f2bf(sqrtf(fmaxf(x * x + y * y + z * z, 1e-8f)));
    }
  }
  __syncthreads();
  // ---- GVP3 (no act) + scatter-add ----
  if (ok) {
    for (int nt = 0; nt < 8; ++nt)
      acc[nt] = wmma_acc_tile(&Xs[w][0][0], XSTR, wpk + W3_OFF + (size_t)nt * K2T * 512, K2T, lane);
    for (int i = lane; i < 256; i += 32) {
      int r = i >> 4, o = i & 15;
      float ox = 0, oy = 0, oz = 0;
      for (int h = 0; h < VI; ++h) {
        float wv = w_v3[o * VI + h];
        ox += wv * VH[w][r][h * 3]; oy += wv * VH[w][r][h * 3 + 1]; oz += wv * VH[w][r][h * 3 + 2];
      }
      VA[w][r][o * 3] = ox; VA[w][r][o * 3 + 1] = oy; VA[w][r][o * 3 + 2] = oz;
    }
    if (lane < 16) atomicAdd(&cnt[DID[w][lane]], 1.0f);
    for (int nt = 0; nt < 8; ++nt) {
#pragma unroll
      for (int q = 0; q < 8; ++q) {
        int m = q + 8 * hi, nc = nt * 16 + row;
        atomicAdd(&agg_s[(size_t)DID[w][m] * SI + nc], acc[nt][q] + b3[nc]);
      }
    }
  }
  __syncthreads();
  if (ok) {
    for (int i = lane; i < 16 * 48; i += 32) {
      int r = i / 48, c = i - r * 48;
      atomicAdd(&agg_v[(size_t)DID[w][r] * 48 + c], VA[w][r][c]);
    }
  }
}

// =========================== node (update) kernel ===========================
#define NKW 2
__global__ __launch_bounds__(64) void node_kernel(
    const float* __restrict__ s, const float* __restrict__ v,
    const float* __restrict__ f1_wh, const float* __restrict__ f1_wsb, const float* __restrict__ f1_wv,
    const float* __restrict__ f2_wh, const float* __restrict__ f2_wsb, const float* __restrict__ f2_wv,
    const float* __restrict__ ln0_g, const float* __restrict__ ln0_b,
    const float* __restrict__ ln1_g, const float* __restrict__ ln1_b,
    const __bf16* __restrict__ wpk,
    const float* __restrict__ agg_s, const float* __restrict__ agg_v, const float* __restrict__ cnt,
    float* __restrict__ out_s, float* __restrict__ out_v) {
  __shared__ float SB[NKW][16][132];
  __shared__ float VB[NKW][16][48];
  __shared__ float VH[NKW][16][96];
  __shared__ float VT[NKW][16][96];
  __shared__ __align__(16) __bf16 X[NKW][16][X2STR];
  __shared__ float w_h1[32 * 16], w_v1[32 * 32], w_h2[32 * 32], w_v2[16 * 32];
  __shared__ float fb1[512], fb2[128], g0[128], be0[128], g1[128], be1[128];
  __shared__ float st[NKW][16][4];

  const int tid = threadIdx.x;
  for (int i = tid; i < 512; i += 64)  { w_h1[i] = f1_wh[i]; fb1[i] = f1_wsb[i]; w_v2[i] = f2_wv[i]; }
  for (int i = tid; i < 1024; i += 64) { w_v1[i] = f1_wv[i]; w_h2[i] = f2_wh[i]; }
  for (int i = tid; i < 128; i += 64) {
    fb2[i] = f2_wsb[i]; g0[i] = ln0_g[i]; be0[i] = ln0_b[i]; g1[i] = ln1_g[i]; be1[i] = ln1_b[i];
  }
  const int w = tid >> 5, lane = tid & 31;
  const int grp = blockIdx.x * NKW + w;
  const bool ok = grp < (NN / 16);
  const int n0 = grp * 16;
  const int row = lane & 15, hi = lane >> 4;
  if (ok && lane < 16) st[w][lane][3] = 1.f / fmaxf(cnt[n0 + lane], 1.0f);
  __syncthreads();

  // scatter-mean finalize + residual
  if (ok) {
    for (int i = lane; i < 16 * SI; i += 32) {
      int r = i >> 7, c = i & 127;
      size_t n = (size_t)(n0 + r);
      SB[w][r][c] = s[n * SI + c] + agg_s[n * SI + c] * st[w][r][3];
    }
    for (int i = lane; i < 16 * 48; i += 32) {
      int r = i / 48, c = i - r * 48;
      size_t n = (size_t)(n0 + r);
      VB[w][r][c] = v[n * 48 + c] + agg_v[n * 48 + c] * st[w][r][3];
    }
  }
  __syncthreads();
  // LN0 stats
  if (ok) {
    if (lane < 16) {
      int r = lane;
      float mu = 0.f;
      for (int c = 0; c < SI; ++c) mu += SB[w][r][c];
      mu *= (1.f / SI);
      float var = 0.f;
      for (int c = 0; c < SI; ++c) { float d = SB[w][r][c] - mu; var += d * d; }
      st[w][r][0] = mu; st[w][r][1] = rsqrtf(var * (1.f / SI) + 1e-5f);
    } else {
      int r = lane - 16;
      float sum = 0.f, cm = 0.f;
      for (int c = 0; c < VI; ++c) {
        float x = VB[w][r][c * 3], y = VB[w][r][c * 3 + 1], z = VB[w][r][c * 3 + 2];
        float n2 = fmaxf(x * x + y * y + z * z, 1e-8f);
        if (n2 > 2e-8f) { sum += n2; cm += 1.f; }
      }
      st[w][r][2] = 1.f / sqrtf(sum / (1e-8f + cm) + 1e-8f);
    }
  }
  __syncthreads();
  // apply LN0 -> s1 (SB + X cols 0..127), v1 (VB)
  if (ok) {
    for (int i = lane; i < 16 * SI; i += 32) {
      int r = i >> 7, c = i & 127;
      float x = (SB[w][r][c] - st[w][r][0]) * st[w][r][1] * g0[c] + be0[c];
      SB[w][r][c] = x;
      X[w][r][c] = f2bf(x);
    }
    for (int i = lane; i < 256; i += 32) {
      int r = i >> 4, c = i & 15;
      float x = VB[w][r][c * 3], y = VB[w][r][c * 3 + 1], z = VB[w][r][c * 3 + 2];
      float n2 = fmaxf(x * x + y * y + z * z, 1e-8f);
      float m = (n2 > 2e-8f) ? st[w][r][2] : 0.f;
      VB[w][r][c * 3] = x * m; VB[w][r][c * 3 + 1] = y * m; VB[w][r][c * 3 + 2] = z * m;
    }
  }
  __syncthreads();
  // FF1 vector hidden (32ch from 16ch)
  if (ok) {
    for (int i = lane; i < 16 * 96; i += 32) {
      int r = i / 96, rem = i - r * 96, h = rem / 3, t = rem - h * 3;
      float a = 0.f;
      for (int c = 0; c < VI; ++c) a += w_h1[h * VI + c] * VB[w][r][c * 3 + t];
      VH[w][r][h * 3 + t] = a;
    }
  }
  __syncthreads();
  if (ok) {
    for (int i = lane; i < 16 * 32; i += 32) {
      int r = i >> 5, h = i & 31;
      float x = VH[w][r][h * 3], y = VH[w][r][h * 3 + 1], z = VH[w][r][h * 3 + 2];
      X[w][r][SI + h] = f2bf(sqrtf(fmaxf(x * x + y * y + z * z, 1e-8f)));
    }
  }
  __syncthreads();
  // FF1 GEMM: 32 tiles -> relu -> X cols 160..671 ; gated v -> VT
  if (ok) {
    for (int nt = 0; nt < 32; ++nt) {
      v8f a1 = wmma_acc_tile(&X[w][0][0], X2STR, wpk + F1_OFF + (size_t)nt * F1KT * 512, F1KT, lane);
#pragma unroll
      for (int q = 0; q < 8; ++q) {
        int m = q + 8 * hi, nc = nt * 16 + row;
        X[w][m][160 + nc] = f2bf(fmaxf(a1[q] + fb1[nc], 0.f));
      }
    }
    for (int i = lane; i < 16 * 32; i += 32) {
      int r = i >> 5, o = i & 31;
      float ox = 0, oy = 0, oz = 0;
      for (int h = 0; h < 32; ++h) {
        float wv = w_v1[o * 32 + h];
        ox += wv * VH[w][r][h * 3]; oy += wv * VH[w][r][h * 3 + 1]; oz += wv * VH[w][r][h * 3 + 2];
      }
      float n = sqrtf(fmaxf(ox * ox + oy * oy + oz * oz, 1e-8f));
      float g = 1.f / (1.f + expf(-n));
      VT[w][r][o * 3] = ox * g; VT[w][r][o * 3 + 1] = oy * g; VT[w][r][o * 3 + 2] = oz * g;
    }
  }
  __syncthreads();
  // FF2 vector hidden ; vn -> X cols 672..703
  if (ok) {
    for (int i = lane; i < 16 * 96; i += 32) {
      int r = i / 96, rem = i - r * 96, h = rem / 3, t = rem - h * 3;
      float a = 0.f;
      for (int c = 0; c < 32; ++c) a += w_h2[h * 32 + c] * VT[w][r][c * 3 + t];
      VH[w][r][h * 3 + t] = a;
    }
  }
  __syncthreads();
  if (ok) {
    for (int i = lane; i < 16 * 32; i += 32) {
      int r = i >> 5, h = i & 31;
      float x = VH[w][r][h * 3], y = VH[w][r][h * 3 + 1], z = VH[w][r][h * 3 + 2];
      X[w][r][672 + h] = f2bf(sqrtf(fmaxf(x * x + y * y + z * z, 1e-8f)));
    }
  }
  __syncthreads();
  // FF2 GEMM + residual ; v residual
  if (ok) {
    for (int nt = 0; nt < 8; ++nt) {
      v8f a2 = wmma_acc_tile(&X[w][0][160], X2STR, wpk + F2_OFF + (size_t)nt * F2KT * 512, F2KT, lane);
#pragma unroll
      for (int q = 0; q < 8; ++q) {
        int m = q + 8 * hi, nc = nt * 16 + row;
        SB[w][m][nc] += a2[q] + fb2[nc];
      }
    }
    for (int i = lane; i < 256; i += 32) {
      int r = i >> 4, o = i & 15;
      float ox = 0, oy = 0, oz = 0;
      for (int h = 0; h < 32; ++h) {
        float wv = w_v2[o * 32 + h];
        ox += wv * VH[w][r][h * 3]; oy += wv * VH[w][r][h * 3 + 1]; oz += wv * VH[w][r][h * 3 + 2];
      }
      VB[w][r][o * 3] += ox; VB[w][r][o * 3 + 1] += oy; VB[w][r][o * 3 + 2] += oz;
    }
  }
  __syncthreads();
  // LN1 stats
  if (ok) {
    if (lane < 16) {
      int r = lane;
      float mu = 0.f;
      for (int c = 0; c < SI; ++c) mu += SB[w][r][c];
      mu *= (1.f / SI);
      float var = 0.f;
      for (int c = 0; c < SI; ++c) { float d = SB[w][r][c] - mu; var += d * d; }
      st[w][r][0] = mu; st[w][r][1] = rsqrtf(var * (1.f / SI) + 1e-5f);
    } else {
      int r = lane - 16;
      float sum = 0.f, cm = 0.f;
      for (int c = 0; c < VI; ++c) {
        float x = VB[w][r][c * 3], y = VB[w][r][c * 3 + 1], z = VB[w][r][c * 3 + 2];
        float n2 = fmaxf(x * x + y * y + z * z, 1e-8f);
        if (n2 > 2e-8f) { sum += n2; cm += 1.f; }
      }
      st[w][r][2] = 1.f / sqrtf(sum / (1e-8f + cm) + 1e-8f);
    }
  }
  __syncthreads();
  // apply LN1 + store outputs
  if (ok) {
    for (int i = lane; i < 16 * SI; i += 32) {
      int r = i >> 7, c = i & 127;
      out_s[(size_t)(n0 + r) * SI + c] =
          (SB[w][r][c] - st[w][r][0]) * st[w][r][1] * g1[c] + be1[c];
    }
    for (int i = lane; i < 256; i += 32) {
      int r = i >> 4, c = i & 15;
      float x = VB[w][r][c * 3], y = VB[w][r][c * 3 + 1], z = VB[w][r][c * 3 + 2];
      float n2 = fmaxf(x * x + y * y + z * z, 1e-8f);
      float m = (n2 > 2e-8f) ? st[w][r][2] : 0.f;
      size_t n = (size_t)(n0 + r);
      out_v[n * 48 + c * 3] = x * m;
      out_v[n * 48 + c * 3 + 1] = y * m;
      out_v[n * 48 + c * 3 + 2] = z * m;
    }
  }
}

// =============================== launcher ===============================
extern "C" void kernel_launch(void* const* d_in, const int* in_sizes, int n_in,
                              void* d_out, int out_size, void* d_ws, size_t ws_size,
                              hipStream_t stream) {
  (void)in_sizes; (void)n_in; (void)out_size; (void)ws_size;
  const float* s      = (const float*)d_in[0];
  const float* v      = (const float*)d_in[1];
  const int*   eidx   = (const int*)d_in[2];
  const float* edge_s = (const float*)d_in[3];
  const float* edge_v = (const float*)d_in[4];
  const float* m1_wh = (const float*)d_in[5];
  const float* m1_wsw= (const float*)d_in[6];
  const float* m1_wsb= (const float*)d_in[7];
  const float* m1_wv = (const float*)d_in[8];
  const float* m2_wh = (const float*)d_in[9];
  const float* m2_wsw= (const float*)d_in[10];
  const float* m2_wsb= (const float*)d_in[11];
  const float* m2_wv = (const float*)d_in[12];
  const float* m3_wh = (const float*)d_in[13];
  const float* m3_wsw= (const float*)d_in[14];
  const float* m3_wsb= (const float*)d_in[15];
  const float* m3_wv = (const float*)d_in[16];
  const float* f1_wh = (const float*)d_in[17];
  const float* f1_wsw= (const float*)d_in[18];
  const float* f1_wsb= (const float*)d_in[19];
  const float* f1_wv = (const float*)d_in[20];
  const float* f2_wh = (const float*)d_in[21];
  const float* f2_wsw= (const float*)d_in[22];
  const float* f2_wsb= (const float*)d_in[23];
  const float* f2_wv = (const float*)d_in[24];
  const float* ln0_g = (const float*)d_in[25];
  const float* ln0_b = (const float*)d_in[26];
  const float* ln1_g = (const float*)d_in[27];
  const float* ln1_b = (const float*)d_in[28];

  __bf16* wpk  = (__bf16*)d_ws;
  float* agg_s = (float*)((char*)d_ws + WPK_BYTES);
  float* agg_v = agg_s + (size_t)NN * SI;
  float* cnt   = agg_v + (size_t)NN * 48;
  float* out_s = (float*)d_out;
  float* out_v = out_s + (size_t)NN * SI;

  hipMemsetAsync(agg_s, 0, (size_t)(NN * SI + NN * 48 + NN) * sizeof(float), stream);

  pack_w<<<(88  * 512) / 256, 256, 0, stream>>>(m1_wsw, wpk + W1_OFF, 128, 321, 11, 8);
  pack_w<<<(40  * 512) / 256, 256, 0, stream>>>(m2_wsw, wpk + W2_OFF, 128, 144, 5, 8);
  pack_w<<<(40  * 512) / 256, 256, 0, stream>>>(m3_wsw, wpk + W3_OFF, 128, 144, 5, 8);
  pack_w<<<(160 * 512) / 256, 256, 0, stream>>>(f1_wsw, wpk + F1_OFF, 512, 160, 5, 32);
  pack_w<<<(136 * 512) / 256, 256, 0, stream>>>(f2_wsw, wpk + F2_OFF, 128, 544, 17, 8);

  edge_kernel<<<(NE / 16 + EKW - 1) / EKW, 128, 0, stream>>>(
      s, v, eidx, edge_s, edge_v,
      m1_wh, m1_wsb, m1_wv, m2_wh, m2_wsb, m2_wv, m3_wh, m3_wsb, m3_wv,
      wpk, agg_s, agg_v, cnt);

  node_kernel<<<(NN / 16 + NKW - 1) / NKW, 64, 0, stream>>>(
      s, v, f1_wh, f1_wsb, f1_wv, f2_wh, f2_wsb, f2_wv,
      ln0_g, ln0_b, ln1_g, ln1_b, wpk, agg_s, agg_v, cnt, out_s, out_v);
}
